// Attention_25460566131147
// MI455X (gfx1250) — matrix operand order repaired and checked by reading, not compiled, not run on hardware
//
#include <hip/hip_runtime.h>

// ---------------------------------------------------------------------------
// Attention layer for MI455X (gfx1250, wave32, WMMA).
// B=4, TQ=T=2048, E=2048, H=16, D=128.
// ~412 GFLOP vs ~300MB traffic => compute bound => bf16 WMMA (fp32 accum),
// async LDS-DMA staging (GLOBAL_LOAD_ASYNC_TO_LDS_B128, ASYNCcnt) with
// double buffering in the GEMMs.
// ---------------------------------------------------------------------------

typedef __bf16 v16bf __attribute__((ext_vector_type(16)));
typedef float  v8f   __attribute__((ext_vector_type(8)));
typedef unsigned int u32x4 __attribute__((ext_vector_type(4)));

union Frag32 { u32x4 q[2]; v16bf v; };   // 32 bytes: one 16-bf16 WMMA operand

static __device__ __forceinline__ unsigned short f2bf(float f) {
    unsigned u = __float_as_uint(f);
    u += 0x7FFFu + ((u >> 16) & 1u);     // round to nearest even
    return (unsigned short)(u >> 16);
}
static __device__ __forceinline__ float bf2f(unsigned short h) {
    return __uint_as_float(((unsigned)h) << 16);
}
static __device__ __forceinline__ v8f wmma_bf16(v16bf a, v16bf b, v8f c) {
    return __builtin_amdgcn_wmma_f32_16x16x32_bf16(false, a, false, b,
                                                   (short)0, c, false, false);
}

// generic->LDS byte offset: flat LDS mapping uses addr[31:0] (ISA 10.2)
static __device__ __forceinline__ unsigned lds_off(const void* p) {
    return (unsigned)(unsigned long long)p;
}

// async DMA: 32B (2 x b128) global -> LDS, no VGPR round trip (ASYNCcnt)
static __device__ __forceinline__ void async_copy32(unsigned lds,
                                                    const void* gptr) {
    asm volatile(
        "global_load_async_to_lds_b128 %0, %1, off\n\t"
        "global_load_async_to_lds_b128 %0, %1, off offset:16"
        :: "v"(lds), "v"(gptr) : "memory");
}
static __device__ __forceinline__ void async_wait0() {
    asm volatile("s_wait_asynccnt 0x0" ::: "memory");
}

// -------------------------- fp32 -> bf16 convert ---------------------------
__global__ void cvt_f32_bf16(const float* __restrict__ in,
                             unsigned short* __restrict__ out, int n) {
    int i = blockIdx.x * blockDim.x + threadIdx.x;
    if (i < n) out[i] = f2bf(in[i]);
}

// ------------------ transpose fp32 weight -> bf16 N-major ------------------
// W: R x C (fp32) -> Wt: C x R (bf16). Block 32x8, tile 32x32.
__global__ void transpose_w(const float* __restrict__ W,
                            unsigned short* __restrict__ Wt, int R, int C) {
    __shared__ float tile[32][33];
    int c0 = blockIdx.x * 32, r0 = blockIdx.y * 32;
    int tx = threadIdx.x, ty = threadIdx.y;
    for (int i = ty; i < 32; i += 8)
        tile[i][tx] = W[(size_t)(r0 + i) * C + c0 + tx];
    __syncthreads();
    for (int i = ty; i < 32; i += 8)
        Wt[(size_t)(c0 + i) * R + r0 + tx] = f2bf(tile[tx][i]);
}

// ------------------------- V transpose (bf16) ------------------------------
// V: [BH][T][D] -> Vt: [BH][D][T]
__global__ void transpose_v(const unsigned short* __restrict__ V,
                            unsigned short* __restrict__ Vt, int T_, int D_) {
    __shared__ unsigned short tile[32][33];
    int bh = blockIdx.z;
    int d0 = blockIdx.x * 32, t0 = blockIdx.y * 32;
    const unsigned short* Vp = V + (size_t)bh * T_ * D_;
    unsigned short* Vtp = Vt + (size_t)bh * T_ * D_;
    int tx = threadIdx.x, ty = threadIdx.y;
    for (int i = ty; i < 32; i += 8)
        tile[i][tx] = Vp[(size_t)(t0 + i) * D_ + d0 + tx];
    __syncthreads();
    for (int i = ty; i < 32; i += 8)
        Vtp[(size_t)(d0 + i) * T_ + t0 + tx] = tile[tx][i];
}

// ------------------------------ RoPE (bf16, in place) ----------------------
// Q: [B][H][T][D]; pos: [B][T]. One thread per (b,h,t,i) with i < D/2.
__global__ void rope_kernel(unsigned short* __restrict__ Q,
                            const int* __restrict__ pos,
                            int Hh, int Tt, int Dd, int n) {
    int idx = blockIdx.x * blockDim.x + threadIdx.x;
    if (idx >= n) return;
    int half = Dd >> 1;
    int i = idx & (half - 1);
    int rest = idx / half;
    int t = rest % Tt;
    int bh = rest / Tt;
    int b = bh / Hh;
    float p = (float)pos[(size_t)b * Tt + t];
    // inv_freq = theta^(-2i/D) = exp(-ln(10000) * 2i/D)
    float freq = __expf(-9.210340372f * (2.0f * (float)i) / (float)Dd);
    float ang = p * freq;
    float s, c;
    __sincosf(ang, &s, &c);
    unsigned short* base = Q + ((size_t)bh * Tt + t) * Dd;
    float x1 = bf2f(base[i]);
    float x2 = bf2f(base[i + half]);
    base[i]        = f2bf(x1 * c - x2 * s);
    base[i + half] = f2bf(x2 * c + x1 * s);
}

// ------------------------------ bf16 WMMA GEMM -----------------------------
// C[M,N] = A[M,K] * Bt[N,K]^T  (A row-major, B pre-transposed to N-major).
// Block tile 128x128, K-step 32, 8 waves: wave grid 4(M) x 2(N),
// each wave: 2x4 tiles of 16x16 => 8 wmma per K-step.
// Staging: double-buffered GLOBAL_LOAD_ASYNC_TO_LDS_B128 (ASYNCcnt), one
// barrier per K-step; next tile's DMA overlaps current tile's WMMAs.
// mode 0: fp32 row-major C.  mode 1: bf16 scatter into [B,H,T,D].
__global__ __launch_bounds__(256)
void gemm_bf16(const unsigned short* __restrict__ A,
               const unsigned short* __restrict__ Bt,
               int M, int N, int K,
               float* __restrict__ Cf,
               unsigned short* __restrict__ Cbf,
               int mode, int Hh, int Dd, int Tt) {
    __shared__ __align__(16) unsigned short As[2][128][40];  // padded rows
    __shared__ __align__(16) unsigned short Bs[2][128][40];

    int tid = threadIdx.x;
    int lane = tid & 31, wave = tid >> 5;
    int lo = lane & 15, hi = lane >> 4;
    int wm = wave & 3, wn = wave >> 2;
    int m0 = blockIdx.y * 128, n0 = blockIdx.x * 128;

    v8f zero = {};
    v8f acc[2][4];
#pragma unroll
    for (int i = 0; i < 2; i++)
#pragma unroll
        for (int j = 0; j < 4; j++) acc[i][j] = zero;

    // 256 threads stage a 128x32 bf16 tile: 2 threads per row, 32B each.
    int lrow = tid >> 1;
    int lcol = (tid & 1) * 16;
    const unsigned short* Aptr = A + (size_t)(m0 + lrow) * K + lcol;
    const unsigned short* Bptr = Bt + (size_t)(n0 + lrow) * K + lcol;
    unsigned ldsA[2], ldsB[2];
    ldsA[0] = lds_off(&As[0][lrow][lcol]);
    ldsA[1] = lds_off(&As[1][lrow][lcol]);
    ldsB[0] = lds_off(&Bs[0][lrow][lcol]);
    ldsB[1] = lds_off(&Bs[1][lrow][lcol]);

    int NK = K >> 5;
    // prologue: kick off tile 0
    async_copy32(ldsA[0], Aptr);
    async_copy32(ldsB[0], Bptr);
    Aptr += 32; Bptr += 32;

    for (int k = 0; k < NK; ++k) {
        int cur = k & 1;
        async_wait0();          // our share of tile k has landed in LDS
        __syncthreads();        // everyone's share landed; buf[next] free
        if (k + 1 < NK) {       // DMA tile k+1 while we compute tile k
            int nxt = cur ^ 1;
            async_copy32(ldsA[nxt], Aptr);
            async_copy32(ldsB[nxt], Bptr);
            Aptr += 32; Bptr += 32;
        }

        Frag32 fa[2], fb[4];
#pragma unroll
        for (int i = 0; i < 2; i++) {
            int r = wm * 32 + i * 16 + lo;                 // A row, K chunks
            fa[i].q[0] = *(const u32x4*)&As[cur][r][8 * hi];
            fa[i].q[1] = *(const u32x4*)&As[cur][r][16 + 8 * hi];
        }
#pragma unroll
        for (int j = 0; j < 4; j++) {
            int r = wn * 64 + j * 16 + lo;                 // B col (N-major)
            fb[j].q[0] = *(const u32x4*)&Bs[cur][r][8 * hi];
            fb[j].q[1] = *(const u32x4*)&Bs[cur][r][16 + 8 * hi];
        }
#pragma unroll
        for (int i = 0; i < 2; i++)
#pragma unroll
            for (int j = 0; j < 4; j++)
                acc[i][j] = wmma_bf16(fa[i].v, fb[j].v, acc[i][j]);
    }

    // Epilogue. C layout: VGPR p -> row p + 8*hi, col = lo.
#pragma unroll
    for (int i = 0; i < 2; i++)
#pragma unroll
        for (int j = 0; j < 4; j++)
#pragma unroll
            for (int p = 0; p < 8; p++) {
                int row = m0 + wm * 32 + i * 16 + p + 8 * hi;
                int col = n0 + wn * 64 + j * 16 + lo;
                float v = acc[i][j][p];
                if (mode == 0) {
                    Cf[(size_t)row * N + col] = v;
                } else {
                    int b = row / Tt, t = row - b * Tt;
                    int h = col / Dd, d = col - h * Dd;
                    Cbf[((size_t)(b * Hh + h) * Tt + t) * Dd + d] = f2bf(v);
                }
            }
}

// --------------------------- flash attention -------------------------------
// One wave per 16 query rows of one (b,h). Online softmax over T in steps of
// 32 kv. QK^T: 8 wmma (4 d-frags x 2 kv subtiles). P bounced via LDS into
// A-fragment layout, then P*V: 8 wmma (8 d-tiles, k=32).
__global__ __launch_bounds__(32)
void attn_kernel(const unsigned short* __restrict__ Q,   // [BH][TQ][D]
                 const unsigned short* __restrict__ Kk,  // [BH][T][D]
                 const unsigned short* __restrict__ Vt,  // [BH][D][T]
                 const unsigned char* __restrict__ mask, // [B][T], 1 = masked
                 unsigned short* __restrict__ Y,         // [B*TQ][H*D] bf16
                 int B_, int H_, int TQ_, int T_, int D_) {
    int bh = blockIdx.x;
    int b = bh / H_, h = bh - b * H_;
    int q0 = blockIdx.y * 16;
    int lane = threadIdx.x;
    int lo = lane & 15, hi = lane >> 4;

    __shared__ __align__(16) unsigned short Pl[16][32];   // P tile, bf16

    const float scale = 0.08838834764831845f;             // 1/sqrt(128)
    const float NEG = -3.0e38f;

    // Preload the 4 Q fragments for this 16-row tile (d = 0..127).
    Frag32 fq[4];
    const unsigned short* Qrow = Q + ((size_t)bh * TQ_ + q0 + lo) * D_;
#pragma unroll
    for (int f = 0; f < 4; f++) {
        fq[f].q[0] = *(const u32x4*)(Qrow + 32 * f + 8 * hi);
        fq[f].q[1] = *(const u32x4*)(Qrow + 32 * f + 16 + 8 * hi);
    }

    v8f zero = {};
    v8f o[8];
#pragma unroll
    for (int dt = 0; dt < 8; dt++) o[dt] = zero;
    float m_s[8], l_s[8], sf[8];
#pragma unroll
    for (int p = 0; p < 8; p++) { m_s[p] = NEG; l_s[p] = 0.0f; }

    const unsigned short* Kbase = Kk + (size_t)bh * T_ * D_;
    const unsigned short* Vbase = Vt + (size_t)bh * D_ * T_;
    const unsigned char* mrow = mask + (size_t)b * T_;

    for (int t0 = 0; t0 < T_; t0 += 32) {
        // ---- scores: Q[16x128] x K^T[128x32] via 2 x 4 wmma ----
        v8f s0 = zero, s1 = zero;
        const unsigned short* K0 = Kbase + (size_t)(t0 + lo) * D_;
        const unsigned short* K1 = Kbase + (size_t)(t0 + 16 + lo) * D_;
#pragma unroll
        for (int f = 0; f < 4; f++) {
            Frag32 fk0, fk1;
            fk0.q[0] = *(const u32x4*)(K0 + 32 * f + 8 * hi);
            fk0.q[1] = *(const u32x4*)(K0 + 32 * f + 16 + 8 * hi);
            fk1.q[0] = *(const u32x4*)(K1 + 32 * f + 8 * hi);
            fk1.q[1] = *(const u32x4*)(K1 + 32 * f + 16 + 8 * hi);
            s0 = wmma_bf16(fq[f].v, fk0.v, s0);
            s1 = wmma_bf16(fq[f].v, fk1.v, s1);
        }

        bool mk0 = mrow[t0 + lo] != 0;
        bool mk1 = mrow[t0 + 16 + lo] != 0;

        // ---- online softmax; stats uniform across each 16-lane group ----
#pragma unroll
        for (int p = 0; p < 8; p++) {
            float x0 = mk0 ? NEG : s0[p] * scale;
            float x1 = mk1 ? NEG : s1[p] * scale;
            float mt = fmaxf(x0, x1);
#pragma unroll
            for (int off = 1; off < 16; off <<= 1)
                mt = fmaxf(mt, __shfl_xor(mt, off, 32));
            float mn = fmaxf(m_s[p], mt);
            sf[p] = __expf(m_s[p] - mn);
            m_s[p] = mn;
            float p0 = mk0 ? 0.0f : __expf(x0 - mn);
            float p1 = mk1 ? 0.0f : __expf(x1 - mn);
            float rs = p0 + p1;
#pragma unroll
            for (int off = 1; off < 16; off <<= 1)
                rs += __shfl_xor(rs, off, 32);
            l_s[p] = l_s[p] * sf[p] + rs;
            // park P in LDS in C layout (row p+8hi, col lo / 16+lo)
            Pl[p + 8 * hi][lo]      = f2bf(p0);
            Pl[p + 8 * hi][16 + lo] = f2bf(p1);
        }

        // Re-read P in A-fragment layout (same-wave LDS ops are in order).
        Frag32 fp;
        fp.q[0] = *(const u32x4*)&Pl[lo][8 * hi];
        fp.q[1] = *(const u32x4*)&Pl[lo][16 + 8 * hi];

        // ---- o += P[16x32] x V[32x128]; rescale running accumulators ----
#pragma unroll
        for (int dt = 0; dt < 8; dt++) {
#pragma unroll
            for (int p = 0; p < 8; p++) o[dt][p] *= sf[p];
            Frag32 fv;
            const unsigned short* Vrow =
                Vbase + (size_t)(dt * 16 + lo) * T_ + t0 + 8 * hi;
            fv.q[0] = *(const u32x4*)(Vrow);
            fv.q[1] = *(const u32x4*)(Vrow + 16);
            o[dt] = wmma_bf16(fp.v, fv.v, o[dt]);
        }
    }

    // ---- epilogue: y = o / l, bf16, row-major [B*TQ][H*D] for final GEMM ----
    int HD = H_ * D_;
#pragma unroll
    for (int dt = 0; dt < 8; dt++)
#pragma unroll
        for (int p = 0; p < 8; p++) {
            int q = q0 + p + 8 * hi;
            float v = o[dt][p] / l_s[p];
            Y[((size_t)b * TQ_ + q) * HD + h * D_ + dt * 16 + lo] = f2bf(v);
        }
}

// ---------------------------------------------------------------------------
extern "C" void kernel_launch(void* const* d_in, const int* in_sizes, int n_in,
                              void* d_out, int out_size, void* d_ws, size_t ws_size,
                              hipStream_t stream) {
    const int B = 4, TQ = 2048, T = 2048, E = 2048, H = 16, D = 128;
    const int HD = H * D;               // 2048
    const int M = B * TQ;               // 8192

    const float* x    = (const float*)d_in[0];
    const float* xall = (const float*)d_in[1];
    const int* posx    = (const int*)d_in[2];
    const int* posxall = (const int*)d_in[3];
    const unsigned char* mask = (const unsigned char*)d_in[4];
    const float* Wq = (const float*)d_in[5];
    const float* Wk = (const float*)d_in[6];
    const float* Wv = (const float*)d_in[7];
    const float* Wo = (const float*)d_in[8];
    float* out = (float*)d_out;

    // workspace carve-up (bf16 buffers)
    size_t off = 0;
    char* ws = (char*)d_ws;
    auto carve = [&](size_t bytes) -> unsigned short* {
        unsigned short* p = (unsigned short*)(ws + off);
        off = (off + bytes + 255) & ~(size_t)255;
        return p;
    };
    unsigned short* xb    = carve((size_t)M * E * 2);
    unsigned short* xallb = carve((size_t)M * E * 2);
    unsigned short* Wtq = carve((size_t)HD * E * 2);
    unsigned short* Wtk = carve((size_t)HD * E * 2);
    unsigned short* Wtv = carve((size_t)HD * E * 2);
    unsigned short* Wto = carve((size_t)E * HD * 2);
    unsigned short* Qb  = carve((size_t)M * HD * 2);
    unsigned short* Kb  = carve((size_t)M * HD * 2);
    unsigned short* Vb  = carve((size_t)M * HD * 2);
    unsigned short* Vtb = carve((size_t)M * HD * 2);
    unsigned short* Yb  = carve((size_t)M * HD * 2);

    // 1) fp32 -> bf16 activations
    int nx = M * E;
    cvt_f32_bf16<<<nx / 256, 256, 0, stream>>>(x, xb, nx);
    cvt_f32_bf16<<<nx / 256, 256, 0, stream>>>(xall, xallb, nx);

    // 2) weights: transpose to N-major + bf16
    dim3 tb(32, 8);
    transpose_w<<<dim3(HD / 32, E / 32), tb, 0, stream>>>(Wq, Wtq, E, HD);
    transpose_w<<<dim3(HD / 32, E / 32), tb, 0, stream>>>(Wk, Wtk, E, HD);
    transpose_w<<<dim3(HD / 32, E / 32), tb, 0, stream>>>(Wv, Wtv, E, HD);
    transpose_w<<<dim3(E / 32, HD / 32), tb, 0, stream>>>(Wo, Wto, HD, E);

    // 3) Q/K/V projections (WMMA GEMM, scatter into [B,H,T,D] bf16)
    dim3 gp(HD / 128, M / 128);
    gemm_bf16<<<gp, 256, 0, stream>>>(xb,    Wtq, M, HD, E, nullptr, Qb, 1, H, D, TQ);
    gemm_bf16<<<gp, 256, 0, stream>>>(xallb, Wtk, M, HD, E, nullptr, Kb, 1, H, D, T);
    gemm_bf16<<<gp, 256, 0, stream>>>(xallb, Wtv, M, HD, E, nullptr, Vb, 1, H, D, T);

    // 4) RoPE on Q and K (in place)
    int nr = B * H * TQ * (D / 2);
    rope_kernel<<<nr / 256, 256, 0, stream>>>(Qb, posx,    H, TQ, D, nr);
    rope_kernel<<<nr / 256, 256, 0, stream>>>(Kb, posxall, H, T,  D, nr);

    // 5) V -> Vt  [BH][D][T]
    transpose_v<<<dim3(D / 32, T / 32, B * H), tb, 0, stream>>>(Vb, Vtb, T, D);

    // 6) flash attention -> Yb bf16 [M][HD]
    attn_kernel<<<dim3(B * H, TQ / 16), 32, 0, stream>>>(Qb, Kb, Vtb, mask, Yb,
                                                         B, H, TQ, T, D);

    // 7) output projection -> fp32 d_out
    gemm_bf16<<<dim3(E / 128, M / 128), 256, 0, stream>>>(Yb, Wto, M, E, HD,
                                                          out, nullptr, 0, H, D, TQ);
}
